// CifPredictorV3_10118942949938
// MI455X (gfx1250) — matrix operand order repaired and checked by reading, not compiled, not run on hardware
//
#include <hip/hip_runtime.h>
#include <hip/hip_bf16.h>

// ---------------------------------------------------------------------------
// CIF predictor for MI455X (gfx1250, wave32, WMMA bf16 16x16x32, f32 accum)
// ---------------------------------------------------------------------------

#define B_   16
#define T_   1000
#define D_   512
#define TT_  32      // t-tile width per block
#define NTT_ 32      // ceil(1000/32)
#define ROWS_ 34     // t0-1 .. t0+32 (conv taps k=0..2)
#define LDB_ 520     // LDS row stride in bf16 elems (512 + 8 pad -> no bank clash)
#define LMAX_ 50

typedef __attribute__((ext_vector_type(8)))  float          v8f;
typedef __attribute__((ext_vector_type(8)))  unsigned short u16x8;
typedef __attribute__((ext_vector_type(16))) __bf16         v16bf;

union FragU { struct { u16x8 lo; u16x8 hi; } p; v16bf v; };

__device__ __forceinline__ unsigned short f2bf(float f) {
  unsigned int u = __float_as_uint(f);
  unsigned int r = u + 0x7FFFu + ((u >> 16) & 1u);   // RNE
  return (unsigned short)(r >> 16);
}
__device__ __forceinline__ float sigmoidf(float x) {
  return 1.0f / (1.0f + __expf(-x));
}

// ---------------------------------------------------------------------------
// Kernel 0: pack conv_w -> bf16 [k][o][i]; w2eff/c2 fold of deconv+out2 proj;
//           zero token_num / token_num2 output slots.
// ---------------------------------------------------------------------------
__global__ __launch_bounds__(256) void prep_kernel(
    const float* __restrict__ conv_w,   // [D][D][3]
    const float* __restrict__ up_w,     // [D][D][5]
    const float* __restrict__ up_b,     // [D]
    const float* __restrict__ out_w,    // [1][D]
    const float* __restrict__ out2_w,   // [1][D]
    const float* __restrict__ out2_b,   // [1]
    unsigned short* __restrict__ wpack, // [3][D][D] bf16
    float* __restrict__ wvec,           // [6][D]  (0: out_w, 1..5: w2eff[:,k])
    float* __restrict__ c2,             // [1]
    float* __restrict__ tnum,           // [B]  (zeroed)
    float* __restrict__ tnum2)          // [B]  (zeroed)
{
  const int NP = 3 * D_ * D_;                    // 786432
  int id = blockIdx.x * 256 + threadIdx.x;
  if (id < NP) {
    int k = id >> 18;                            // / (512*512)
    int rem = id & (D_ * D_ - 1);
    int o = rem >> 9, i = rem & (D_ - 1);
    wpack[id] = f2bf(conv_w[(o * D_ + i) * 3 + k]);
  } else if (id < NP + D_) {
    int o = id - NP;
    wvec[o] = out_w[o];
  } else if (id < NP + D_ + 5 * D_) {
    int j = id - NP - D_;
    int i = j / 5, k = j % 5;
    float s = 0.f;
    for (int o = 0; o < D_; ++o) s += up_w[(i * D_ + o) * 5 + k] * out2_w[o];
    wvec[(1 + k) * D_ + i] = s;
  } else if (id == NP + 6 * D_) {
    float s = 0.f;
    for (int o = 0; o < D_; ++o) s += up_b[o] * out2_w[o];
    c2[0] = s + out2_b[0];
  } else if (id < NP + 6 * D_ + 1 + 2 * B_) {
    int j = id - (NP + 6 * D_ + 1);
    if (j < B_) tnum[j] = 0.f; else tnum2[j - B_] = 0.f;
  }
}

// ---------------------------------------------------------------------------
// Kernel 1: conv GEMM (bf16 WMMA, f32 acc) with fused bias+relu+6-dot epilogue.
// Block = (b, t-tile of 32). 8 waves: wave w -> t-half (w&1), o-group (w>>1),
// 8 o-tiles each -> full O=512 x 32 t per block. Never stores conv output.
// ---------------------------------------------------------------------------
__global__ __launch_bounds__(256) void conv_dot_kernel(
    const float* __restrict__ hidden,   // [B][T][D]
    const float* __restrict__ mask,     // [B][1][T]
    const float* __restrict__ conv_b,   // [D]
    const float* __restrict__ out_b,    // [1]
    const unsigned short* __restrict__ wpack, // [3][D][D] bf16
    const float* __restrict__ wvec,     // [6][D]
    const float* __restrict__ c2p,      // [1]
    float* __restrict__ alphas_raw,     // [B][T]  (d_out alphas slot, raw)
    float* __restrict__ tnum,           // [B] atomic
    float* __restrict__ tnum2)          // [B] atomic
{
  const int b  = blockIdx.x >> 5;       // / NTT_
  const int tt = blockIdx.x & 31;
  const int t0 = tt * TT_;
  const int tid = threadIdx.x;

  __shared__ unsigned short Bt[ROWS_ * LDB_];   // bf16 hidden tile
  __shared__ float sdot[TT_][8];                // 6 used

  ((float*)sdot)[tid] = 0.f;                    // 256 == 32*8

  // stage hidden rows t0-1 .. t0+32 as bf16 (zero pad outside [0,T))
  for (int e = tid; e < ROWS_ * (D_ / 4); e += 256) {
    int row = e >> 7;
    int c4  = (e & 127) << 2;
    int gt  = t0 - 1 + row;
    float4 v = make_float4(0.f, 0.f, 0.f, 0.f);
    if (gt >= 0 && gt < T_)
      v = *reinterpret_cast<const float4*>(hidden + ((size_t)b * T_ + gt) * D_ + c4);
    unsigned short* p = &Bt[row * LDB_ + c4];
    p[0] = f2bf(v.x); p[1] = f2bf(v.y); p[2] = f2bf(v.z); p[3] = f2bf(v.w);
  }
  __syncthreads();

  const int lane  = tid & 31;
  const int wid   = tid >> 5;
  const int thalf = wid & 1;
  const int og    = wid >> 1;
  const int n     = lane & 15;
  const int kbA   = (lane >> 4) * 8;    // A: per-lane-half K base (ISA 7.12.2)
  const int kbB   = (lane >> 4) * 8;    // B: per-lane-half K base
  const int tl    = thalf * 16 + n;     // local t column owned by this lane

  float dacc[6] = {0.f, 0.f, 0.f, 0.f, 0.f, 0.f};

  for (int jj = 0; jj < 8; ++jj) {
    const int o0 = (og + 4 * jj) * 16;
    v8f acc;
#pragma unroll
    for (int r = 0; r < 8; ++r) acc[r] = 0.f;

    for (int k = 0; k < 3; ++k) {
      const unsigned short* Ap = wpack + ((size_t)(k * D_ + (o0 + n)) * D_) + kbA;
      const unsigned short* Bp = &Bt[(tl + k) * LDB_ + kbB];
#pragma unroll 4
      for (int c = 0; c < 16; ++c) {            // i-chunks of 32
        FragU a, bf;
        a.p.lo  = *reinterpret_cast<const u16x8*>(Ap);        // K kbA..+7
        a.p.hi  = *reinterpret_cast<const u16x8*>(Ap + 16);   // K kbA+16..+23
        bf.p.lo = *reinterpret_cast<const u16x8*>(Bp);        // K kbB..+7
        bf.p.hi = *reinterpret_cast<const u16x8*>(Bp + 16);   // K kbB+16..+23
        acc = __builtin_amdgcn_wmma_f32_16x16x32_bf16(
            false, a.v, false, bf.v, (short)0, acc, false, false);
        Ap += 32; Bp += 32;
      }
    }
    // fused epilogue: relu(acc + conv_b) dotted with 6 weight vectors
    const int obase = o0 + kbA;   // C layout: vgpr r -> M = r + 8*(lane>=16)
#pragma unroll
    for (int r = 0; r < 8; ++r) {
      const int o = obase + r;
      float v = acc[r] + conv_b[o];
      v = v > 0.f ? v : 0.f;
#pragma unroll
      for (int dv = 0; dv < 6; ++dv)
        dacc[dv] += v * wvec[dv * D_ + o];
    }
  }
#pragma unroll
  for (int dv = 0; dv < 6; ++dv)
    atomicAdd(&sdot[tl][dv], dacc[dv]);
  __syncthreads();

  if (tid < TT_) {
    const int t = t0 + tid;
    if (t < T_) {
      const float m  = mask[(size_t)b * T_ + t];
      const float c2 = c2p[0];
      // SMOOTH=1, NOISE=0 -> relu(sigmoid(x)) == sigmoid(x)
      const float a = sigmoidf(sdot[tid][0] + out_b[0]) * m;
      alphas_raw[(size_t)b * T_ + t] = a;
      float s2 = 0.f;
#pragma unroll
      for (int k = 0; k < 5; ++k) s2 += sigmoidf(sdot[tid][1 + k] + c2);
      s2 *= m;
      atomicAdd(&tnum[b], a);
      atomicAdd(&tnum2[b], s2);
    }
  }
}

// ---------------------------------------------------------------------------
// Kernel 2: CIF. One block per batch. Lane 0: exact sequential scan (from LDS)
// producing per-t weights/segments + cif_peak + scaled alphas. Then 256
// threads stream hidden (float2/lane) replaying the frame recurrence.
// ---------------------------------------------------------------------------
__global__ __launch_bounds__(256) void cif_kernel(
    const float* __restrict__ hidden,   // [B][T][D]
    const float* __restrict__ tll,      // [B]
    float* __restrict__ alphas,         // [B][T] in: raw, out: scaled
    float* __restrict__ peak,           // [B][T]
    float* __restrict__ ae,             // [B][50][D]
    const float* __restrict__ tnum)     // [B]
{
  const int b   = blockIdx.x;
  const int tid = threadIdx.x;

  __shared__ float sal[T_], swc[T_], swr[T_];
  __shared__ int   sseg[T_];
  __shared__ int   snf;

  for (int t = tid; t < T_; t += 256) sal[t] = alphas[(size_t)b * T_ + t];
  __syncthreads();

  if (tid == 0) {
    const float scale = tll[b] / tnum[b];
    float integ = 0.f; int seg = 0;
    for (int t = 0; t < T_; ++t) {
      const float alpha = sal[t] * scale;
      alphas[(size_t)b * T_ + t] = alpha;          // scaled output
      const float dist = 1.0f - integ;
      integ += alpha;
      const bool fire = (integ >= 1.0f);           // THRESHOLD
      peak[(size_t)b * T_ + t] = integ;            // pre-reset
      const float cur = fire ? dist : alpha;
      swc[t]  = cur;
      swr[t]  = fire ? (alpha - cur) : 0.f;
      sseg[t] = seg | (fire ? (1 << 30) : 0);
      if (fire) { integ -= 1.0f; ++seg; }
    }
    snf = seg;
  }
  __syncthreads();

  const float2* H = reinterpret_cast<const float2*>(hidden);
  float2* AE2 = reinterpret_cast<float2*>(ae);
  const size_t base = (size_t)b * T_ * (D_ / 2);
  float2 fr; fr.x = 0.f; fr.y = 0.f;

  for (int t = 0; t < T_; ++t) {
    const float wc = swc[t];
    const int   u  = sseg[t];
    const float2 h = H[base + (size_t)t * (D_ / 2) + tid];
    fr.x += wc * h.x; fr.y += wc * h.y;
    if (u & (1 << 30)) {                            // uniform branch
      const int s = u & 0x3FFFFFFF;
      if (s < LMAX_)
        AE2[((size_t)b * LMAX_ + s) * (D_ / 2) + tid] = fr;
      const float wr = swr[t];
      fr.x = wr * h.x; fr.y = wr * h.y;
    }
  }
  int nf = snf; if (nf > LMAX_) nf = LMAX_;
  float2 z; z.x = 0.f; z.y = 0.f;
  for (int s = nf; s < LMAX_; ++s)
    AE2[((size_t)b * LMAX_ + s) * (D_ / 2) + tid] = z;
}

// ---------------------------------------------------------------------------
extern "C" void kernel_launch(void* const* d_in, const int* in_sizes, int n_in,
                              void* d_out, int out_size, void* d_ws, size_t ws_size,
                              hipStream_t stream) {
  (void)in_sizes; (void)n_in; (void)out_size; (void)ws_size;

  const float* hidden = (const float*)d_in[0];
  const float* mask   = (const float*)d_in[1];
  const float* tll    = (const float*)d_in[2];
  const float* conv_w = (const float*)d_in[3];
  const float* conv_b = (const float*)d_in[4];
  const float* up_w   = (const float*)d_in[5];
  const float* up_b   = (const float*)d_in[6];
  const float* out_w  = (const float*)d_in[7];
  const float* out_b  = (const float*)d_in[8];
  const float* out2_w = (const float*)d_in[9];
  const float* out2_b = (const float*)d_in[10];

  // d_out: acoustic_embeds [16,50,512] | token_num [16] | alphas [16,1000]
  //        | cif_peak [16,1000] | token_num2 [16]
  float* AE  = (float*)d_out;
  float* TN  = AE + B_ * LMAX_ * D_;
  float* AL  = TN + B_;
  float* PK  = AL + B_ * T_;
  float* TN2 = PK + B_ * T_;

  unsigned short* wpack = (unsigned short*)d_ws;                 // 1.5 MB
  float* wvec = (float*)((char*)d_ws + (size_t)3 * D_ * D_ * sizeof(unsigned short));
  float* c2   = wvec + 6 * D_;

  const int prep_items = 3 * D_ * D_ + 6 * D_ + 1 + 2 * B_;
  prep_kernel<<<(prep_items + 255) / 256, 256, 0, stream>>>(
      conv_w, up_w, up_b, out_w, out2_w, out2_b, wpack, wvec, c2, TN, TN2);

  conv_dot_kernel<<<B_ * NTT_, 256, 0, stream>>>(
      hidden, mask, conv_b, out_b, wpack, wvec, c2, AL, TN, TN2);

  cif_kernel<<<B_, 256, 0, stream>>>(hidden, tll, AL, PK, AE, TN);
}
